// MultiHeadAttentionPool3D_37658273251383
// MI455X (gfx1250) — compile-verified, matrix operand order checked by reading, not run on hardware
//
#include <hip/hip_runtime.h>
#include <hip/hip_bf16.h>
#include <cstdint>

// ---------------------------------------------------------------------------
// MultiHeadAttentionPool3D for gfx1250 (MI455X), wave32 + WMMA bf16.
// B=4, C=256, S=32768, F=512, H=8, hd=64, Q=4.  r = h*4+q in [0,32).
// ---------------------------------------------------------------------------

#define S_TOT   32768
#define C_IN    256
#define NROWS   32          // H*Q query-rows
#define SCALE   0.125f      // hd^-0.5

typedef __attribute__((ext_vector_type(16))) __bf16 v16bf;
typedef __attribute__((ext_vector_type(8)))  float  v8f;

union V16 { uint32_t u[8]; uint4 q[2]; v16bf v; };

static __device__ __forceinline__ uint16_t f2bf(float f) {
    uint32_t u = __float_as_uint(f);
    uint32_t r = u + 0x7FFFu + ((u >> 16) & 1u);   // round-to-nearest-even
    return (uint16_t)(r >> 16);
}
static __device__ __forceinline__ uint32_t pk2bf(float lo, float hi) {
    return (uint32_t)f2bf(lo) | ((uint32_t)f2bf(hi) << 16);
}
static __device__ __forceinline__ v8f wmma_bf16(const V16& a, const V16& b, v8f c) {
    return __builtin_amdgcn_wmma_f32_16x16x32_bf16(
        false, a.v, false, b.v, (short)0, c, false, false);
}

// ---------------------------------------------------------------------------
// K1: fold Wk into queries.  qW[r][c] = sum_d q[q,h,d]*Wk[h*64+d][c]  (bf16)
//     qb[r]  = sum_d q[q,h,d]*bk[h*64+d]
// ---------------------------------------------------------------------------
__global__ void k1_prep(const float* __restrict__ queries,
                        const float* __restrict__ Wk,
                        const float* __restrict__ bk,
                        uint16_t* __restrict__ qWbf,
                        float* __restrict__ qb) {
    const int t = threadIdx.x;              // 256 threads, t == channel c
    for (int r = 0; r < NROWS; ++r) {
        const int h = r >> 2, q = r & 3;
        float acc = 0.f;
        #pragma unroll 8
        for (int d = 0; d < 64; ++d)
            acc += queries[q * 512 + h * 64 + d] * Wk[(h * 64 + d) * C_IN + t];
        qWbf[r * C_IN + t] = f2bf(acc);
    }
    if (t < NROWS) {
        const int h = t >> 2, q = t & 3;
        float acc = 0.f;
        for (int d = 0; d < 64; ++d)
            acc += queries[q * 512 + h * 64 + d] * bk[h * 64 + d];
        qb[t] = acc;
    }
}

// ---------------------------------------------------------------------------
// K2: scores P[b][r][s] = SCALE*(qW[r,:]·x[b,:,s] + qb[r])
// GEMM M=32 N=32768(per b) K=256 via v_wmma_f32_16x16x32_bf16.
// Block = 8 waves, covers 1024 s; x staged through LDS as bf16 [s][c].
// ---------------------------------------------------------------------------
__global__ __launch_bounds__(256) void k2_scores(
        const float* __restrict__ x,
        const uint16_t* __restrict__ qWbf,
        const float* __restrict__ qb,
        float* __restrict__ P) {
    __shared__ uint16_t ldsX[128 * C_IN];   // 64 KB, [s_local][c] bf16

    const int b   = blockIdx.y;
    const int sBk = blockIdx.x * 1024;
    const int t   = threadIdx.x;
    const int lane = t & 31, wave = t >> 5;
    const int hs  = lane >> 4;              // half-wave select
    const int l16 = lane & 15;
    const int koffLane = hs ? 8 : 0;        // ISA 16-bit A/B fragment layout

    // A fragments (qW), register-resident: 2 M-tiles x 8 K-chunks
    V16 A[2][8];
    #pragma unroll
    for (int m = 0; m < 2; ++m) {
        const int row = m * 16 + l16;
        #pragma unroll
        for (int kk = 0; kk < 8; ++kk) {
            const int cb = kk * 32 + koffLane;
            A[m][kk].q[0] = *reinterpret_cast<const uint4*>(qWbf + row * C_IN + cb);
            A[m][kk].q[1] = *reinterpret_cast<const uint4*>(qWbf + row * C_IN + cb + 16);
        }
    }

    const float* xb = x + (size_t)b * C_IN * S_TOT;
    float* Pb = P + (size_t)b * NROWS * S_TOT;

    for (int sub = 0; sub < 8; ++sub) {
        const int s0 = sBk + sub * 128;
        __syncthreads();
        // cooperative load: x[256c][128s] f32 -> LDS bf16 transposed [s][c]
        #pragma unroll 4
        for (int r = 0; r < 32; ++r) {
            const int idx = r * 256 + t;          // 0..8191
            const int c   = idx >> 5;             // row (channel)
            const int p4  = idx & 31;             // float4 index along s
            const float4 v = *reinterpret_cast<const float4*>(
                xb + (size_t)c * S_TOT + s0 + p4 * 4);
            const int sl = p4 * 4;
            ldsX[(sl + 0) * C_IN + c] = f2bf(v.x);
            ldsX[(sl + 1) * C_IN + c] = f2bf(v.y);
            ldsX[(sl + 2) * C_IN + c] = f2bf(v.z);
            ldsX[(sl + 3) * C_IN + c] = f2bf(v.w);
        }
        __syncthreads();

        const int sl = wave * 16 + l16;           // this lane's N column
        v8f acc0 = {}; v8f acc1 = {};
        #pragma unroll
        for (int kk = 0; kk < 8; ++kk) {
            V16 Bv;
            const int cb = kk * 32 + koffLane;
            Bv.q[0] = *reinterpret_cast<const uint4*>(&ldsX[sl * C_IN + cb]);
            Bv.q[1] = *reinterpret_cast<const uint4*>(&ldsX[sl * C_IN + cb + 16]);
            acc0 = wmma_bf16(A[0][kk], Bv, acc0);
            acc1 = wmma_bf16(A[1][kk], Bv, acc1);
        }
        const int sg = s0 + sl;
        #pragma unroll
        for (int j = 0; j < 8; ++j) {
            const int r0 = j + (hs ? 8 : 0);
            const int r1 = 16 + j + (hs ? 8 : 0);
            Pb[(size_t)r0 * S_TOT + sg] = (acc0[j] + qb[r0]) * SCALE;
            Pb[(size_t)r1 * S_TOT + sg] = (acc1[j] + qb[r1]) * SCALE;
        }
    }
}

// ---------------------------------------------------------------------------
// K3: per-row softmax stats; write unnormalized exp weights as bf16.
// One block per (b,r) row.
// ---------------------------------------------------------------------------
__global__ __launch_bounds__(256) void k3_softmax(
        const float* __restrict__ P,
        uint16_t* __restrict__ Ebf,
        float* __restrict__ invsum) {
    __shared__ float red[256];
    const int row = blockIdx.x;               // b*32 + r
    const float* Pr = P + (size_t)row * S_TOT;
    uint16_t* Er = Ebf + (size_t)row * S_TOT;
    const int t = threadIdx.x;

    float m = -3.4e38f;
    for (int i = t; i < S_TOT; i += 256) m = fmaxf(m, Pr[i]);
    red[t] = m; __syncthreads();
    for (int o = 128; o > 0; o >>= 1) { if (t < o) red[t] = fmaxf(red[t], red[t + o]); __syncthreads(); }
    const float M = red[0]; __syncthreads();

    float s = 0.f;
    for (int i = t; i < S_TOT; i += 256) {
        const float e = __expf(Pr[i] - M);
        Er[i] = f2bf(e);
        s += e;
    }
    red[t] = s; __syncthreads();
    for (int o = 128; o > 0; o >>= 1) { if (t < o) red[t] += red[t + o]; __syncthreads(); }
    if (t == 0) invsum[row] = 1.f / red[0];
}

// ---------------------------------------------------------------------------
// K4: pooled[b][r][c] = invsum[r] * sum_s E[r,s]*x[c,s]
// GEMM M=32 N=256 K=32768 via WMMA; K split 4-ways across waves.
// Block = (c-tile of 16, batch), 8 waves: wave w -> m = w&1, kq = w>>1.
// ---------------------------------------------------------------------------
__global__ __launch_bounds__(256) void k4_pooled(
        const float* __restrict__ x,
        const uint16_t* __restrict__ Ebf,
        const float* __restrict__ invsum,
        float* __restrict__ pooled) {
    __shared__ float part[8][32][8];
    const int b  = blockIdx.y;
    const int c0 = blockIdx.x * 16;
    const int t = threadIdx.x, lane = t & 31, wave = t >> 5;
    const int hs = lane >> 4, l16 = lane & 15;
    const int koffLane = hs ? 8 : 0;
    const int m = wave & 1, kq = wave >> 1;

    const float* xb = x + (size_t)b * C_IN * S_TOT;
    const uint16_t* Eb = Ebf + (size_t)b * NROWS * S_TOT;

    const int arow = m * 16 + l16;            // A row this lane reads
    const int bcol = c0 + l16;                // B column this lane reads
    v8f acc = {};
    for (int s0 = kq * 8192; s0 < (kq + 1) * 8192; s0 += 32) {
        V16 Av;
        Av.q[0] = *reinterpret_cast<const uint4*>(Eb + (size_t)arow * S_TOT + s0 + koffLane);
        Av.q[1] = *reinterpret_cast<const uint4*>(Eb + (size_t)arow * S_TOT + s0 + koffLane + 16);

        const float* xr = xb + (size_t)bcol * S_TOT + s0 + koffLane;
        const float4 f0 = *reinterpret_cast<const float4*>(xr);
        const float4 f1 = *reinterpret_cast<const float4*>(xr + 4);
        const float4 f2 = *reinterpret_cast<const float4*>(xr + 16);
        const float4 f3 = *reinterpret_cast<const float4*>(xr + 20);
        V16 Bv;
        Bv.u[0] = pk2bf(f0.x, f0.y); Bv.u[1] = pk2bf(f0.z, f0.w);
        Bv.u[2] = pk2bf(f1.x, f1.y); Bv.u[3] = pk2bf(f1.z, f1.w);
        Bv.u[4] = pk2bf(f2.x, f2.y); Bv.u[5] = pk2bf(f2.z, f2.w);
        Bv.u[6] = pk2bf(f3.x, f3.y); Bv.u[7] = pk2bf(f3.z, f3.w);

        acc = wmma_bf16(Av, Bv, acc);
    }
    #pragma unroll
    for (int j = 0; j < 8; ++j) part[wave][lane][j] = acc[j];
    __syncthreads();
    if (wave < 2) {
        const int mm = wave;
        #pragma unroll
        for (int j = 0; j < 8; ++j) {
            const float tot = part[mm][lane][j] + part[2 + mm][lane][j]
                            + part[4 + mm][lane][j] + part[6 + mm][lane][j];
            const int rowg = mm * 16 + j + (hs ? 8 : 0);
            pooled[((size_t)b * NROWS + rowg) * C_IN + (c0 + l16)] =
                tot * invsum[b * NROWS + rowg];
        }
    }
}

// ---------------------------------------------------------------------------
// K5: attended = Wv*pooled + bv ; reorder to multi[q*512+h*64+d];
//     out = multi·Wo^T + bo ; LayerNorm.  One block per batch.
// ---------------------------------------------------------------------------
__global__ __launch_bounds__(512) void k5_final(
        const float* __restrict__ pooled,
        const float* __restrict__ Wv, const float* __restrict__ bv,
        const float* __restrict__ Wo, const float* __restrict__ bo,
        const float* __restrict__ gamma, const float* __restrict__ beta,
        float* __restrict__ out) {
    __shared__ float multi[2048];
    __shared__ float red[512];
    const int b = blockIdx.x, t = threadIdx.x;   // t = h*64+d
    const int h = t >> 6;

    for (int q = 0; q < 4; ++q) {
        const int r = h * 4 + q;
        const float* pr = pooled + ((size_t)b * NROWS + r) * C_IN;
        float acc = bv[t];
        #pragma unroll 4
        for (int c = 0; c < C_IN; ++c) acc += Wv[t * C_IN + c] * pr[c];
        multi[q * 512 + t] = acc;
    }
    __syncthreads();

    float acc = bo[t];
    const float* wo = Wo + (size_t)t * 2048;
    #pragma unroll 4
    for (int j = 0; j < 2048; ++j) acc += multi[j] * wo[j];

    red[t] = acc; __syncthreads();
    for (int o = 256; o > 0; o >>= 1) { if (t < o) red[t] += red[t + o]; __syncthreads(); }
    const float mu = red[0] * (1.f / 512.f); __syncthreads();
    const float d = acc - mu;
    red[t] = d * d; __syncthreads();
    for (int o = 256; o > 0; o >>= 1) { if (t < o) red[t] += red[t + o]; __syncthreads(); }
    const float var = red[0] * (1.f / 512.f);
    out[b * 512 + t] = d * rsqrtf(var + 1e-5f) * gamma[t] + beta[t];
}

// ---------------------------------------------------------------------------
extern "C" void kernel_launch(void* const* d_in, const int* in_sizes, int n_in,
                              void* d_out, int out_size, void* d_ws, size_t ws_size,
                              hipStream_t stream) {
    const float* x       = (const float*)d_in[0];
    const float* queries = (const float*)d_in[1];
    const float* Wk      = (const float*)d_in[2];
    const float* bk      = (const float*)d_in[3];
    const float* Wv      = (const float*)d_in[4];
    const float* bv      = (const float*)d_in[5];
    const float* Wo      = (const float*)d_in[6];
    const float* bo      = (const float*)d_in[7];
    const float* gamma   = (const float*)d_in[8];
    const float* beta    = (const float*)d_in[9];
    float* out = (float*)d_out;

    char* ws = (char*)d_ws;
    uint16_t* qWbf   = (uint16_t*)(ws + 0);                      // 16 KB
    float*    qb     = (float*)   (ws + 16384);                  // 128 B
    float*    invsum = (float*)   (ws + 16896);                  // 512 B
    float*    pooled = (float*)   (ws + 32768);                  // 128 KB
    float*    P      = (float*)   (ws + 262144);                 // 16.8 MB
    uint16_t* Ebf    = (uint16_t*)(ws + 262144 + 16777216);      // 8.4 MB

    k1_prep   <<<1,             256, 0, stream>>>(queries, Wk, bk, qWbf, qb);
    k2_scores <<<dim3(32, 4),   256, 0, stream>>>(x, qWbf, qb, P);
    k3_softmax<<<128,           256, 0, stream>>>(P, Ebf, invsum);
    k4_pooled <<<dim3(16, 4),   256, 0, stream>>>(x, Ebf, invsum, pooled);
    k5_final  <<<4,             512, 0, stream>>>(pooled, Wv, bv, Wo, bo, gamma, beta, out);
}
